// Gumbel_Sinkhorn_Transformer_Network_39539468927343
// MI455X (gfx1250) — matrix operand order repaired and checked by reading, not compiled, not run on hardware
//
#include <hip/hip_runtime.h>

#define N_TOK  256
#define DMODEL 512
#define NHEAD  8
#define DHEAD  64
#define NLAYER 4
#define DFF    2048
#define SEQ    257
#define SPAD   288     // multiple of 32 so WMMA K-loops are exact
#define NITER  20
#define LN_EPS 1e-5f

typedef __attribute__((ext_vector_type(16))) _Float16 v16h;
typedef __attribute__((ext_vector_type(8)))  _Float16 v8h;
typedef __attribute__((ext_vector_type(4)))  _Float16 v4h;
typedef __attribute__((ext_vector_type(8)))  float    v8f;

// ---------------- WMMA fragment helpers (ISA 7.12.2 layouts, wave32) --------
// All operands are f16 in memory, contiguous along K -> 128-bit loads.

// A fragment: 16x32 (MxK) slice of row-major f16 A.
// lane<16: M=lane, halves 0..7 -> K=k0..k0+7, halves 8..15 -> K=k0+16..k0+23
__device__ __forceinline__ v16h frag_a16(const _Float16* __restrict__ A, int lda,
                                         int row0, int k0, int lane) {
  const int g = lane >> 4, m = lane & 15;
  const _Float16* p = A + (size_t)(row0 + m) * lda + k0 + 8 * g;
  const v8h lo = *(const v8h*)p;
  const v8h hi = *(const v8h*)(p + 16);
  return __builtin_shufflevector(lo, hi, 0, 1, 2, 3, 4, 5, 6, 7,
                                 8, 9, 10, 11, 12, 13, 14, 15);
}

// B fragment where B = W^T, W row-major f16 (N x K): two runs of 8 contiguous halves, 16 apart.
// lane<16: N=lane, halves 0..7 -> K=k0..k0+7, halves 8..15 -> K=k0+16..k0+23
__device__ __forceinline__ v16h frag_bt16(const _Float16* __restrict__ W, int ldw,
                                          int k0, int col0, int lane) {
  const int g = lane >> 4, n = lane & 15;
  const _Float16* p = W + (size_t)(col0 + n) * ldw + k0 + 8 * g;
  const v8h lo = *(const v8h*)p;
  const v8h hi = *(const v8h*)(p + 16);
  return __builtin_shufflevector(lo, hi, 0, 1, 2, 3, 4, 5, 6, 7,
                                 8, 9, 10, 11, 12, 13, 14, 15);
}

__device__ __forceinline__ v8f wmma_step(v16h a, v16h b, v8f c) {
  return __builtin_amdgcn_wmma_f32_16x16x32_f16(false, a, false, b,
                                                (short)0, c, false, false);
}

// ---------------- precision conversion / layout kernels --------------------

// straight f32 -> f16 (4 elements/thread, vectorized)
__global__ void k_cvt(const float* __restrict__ in, _Float16* __restrict__ out, int n4) {
  const int i = blockIdx.x * 256 + threadIdx.x;
  if (i >= n4) return;
  const float4 v = ((const float4*)in)[i];
  v4h o;
  o[0] = (_Float16)v.x; o[1] = (_Float16)v.y;
  o[2] = (_Float16)v.z; o[3] = (_Float16)v.w;
  ((v4h*)out)[i] = o;
}

// out[c][r] = (f16) in[r][c]; rows, cols multiples of 32; LDS 32x33 tile
__global__ void k_tr32(const float* __restrict__ in, _Float16* __restrict__ out,
                       int rows, int cols) {
  __shared__ _Float16 tile[32][33];
  const int tx = threadIdx.x & 31;
  const int ty = threadIdx.x >> 5;                 // 0..7
  const int r0 = blockIdx.y * 32, c0 = blockIdx.x * 32;
#pragma unroll
  for (int i = 0; i < 4; ++i) {
    const int r = ty + i * 8;
    tile[r][tx] = (_Float16)in[(size_t)(r0 + r) * cols + c0 + tx];
  }
  __syncthreads();
#pragma unroll
  for (int i = 0; i < 4; ++i) {
    const int c = ty + i * 8;
    out[(size_t)(c0 + c) * rows + r0 + tx] = tile[tx][c];
  }
}

// f16 -> f16 transpose with input leading dimension (for the V block of QKV)
__global__ void k_tr16(const _Float16* __restrict__ in, int ldin,
                       _Float16* __restrict__ out, int rows, int cols) {
  __shared__ _Float16 tile[32][33];
  const int tx = threadIdx.x & 31;
  const int ty = threadIdx.x >> 5;
  const int r0 = blockIdx.y * 32, c0 = blockIdx.x * 32;
#pragma unroll
  for (int i = 0; i < 4; ++i) {
    const int r = ty + i * 8;
    tile[r][tx] = in[(size_t)(r0 + r) * ldin + c0 + tx];
  }
  __syncthreads();
#pragma unroll
  for (int i = 0; i < 4; ++i) {
    const int c = ty + i * 8;
    out[(size_t)(c0 + c) * rows + r0 + tx] = tile[tx][c];
  }
}

// ---------------- GEMM kernels (one wave = one 16x16 tile) -----------------

// src[1..256] = x16(256x64) @ WembT^T + emb_b + pos ; writes f32 + f16 copies
__global__ void k_embed(const _Float16* __restrict__ x16, const _Float16* __restrict__ WembT,
                        const float* __restrict__ emb_b, const float* __restrict__ pos,
                        float* __restrict__ srcf, _Float16* __restrict__ src16) {
  const int wave = (blockIdx.x * blockDim.x + threadIdx.x) >> 5;
  const int lane = threadIdx.x & 31;
  const int MT = N_TOK / 16, NT = DMODEL / 16;
  if (wave >= MT * NT) return;
  const int tm = wave % MT, tn = wave / MT;
  const int row0 = tm * 16, col0 = tn * 16;
  v8f c = {};
#pragma unroll
  for (int k0 = 0; k0 < 64; k0 += 32) {
    c = wmma_step(frag_a16(x16, 64, row0, k0, lane),
                  frag_bt16(WembT, 64, k0, col0, lane), c);
  }
  const int g = lane >> 4, n = lane & 15;
#pragma unroll
  for (int r = 0; r < 8; ++r) {
    const int row = row0 + r + 8 * g;          // token 0..255 -> src row+1
    const int col = col0 + n;
    const float v = c[r] + emb_b[col] + pos[(size_t)(row + 1) * DMODEL + col];
    srcf [(size_t)(row + 1) * DMODEL + col] = v;
    src16[(size_t)(row + 1) * DMODEL + col] = (_Float16)v;
  }
}

// row 0 = cls + pos[0]; rows 257..287 = 0  (both copies)
__global__ void k_edges(const float* __restrict__ cls, const float* __restrict__ pos,
                        float* __restrict__ srcf, _Float16* __restrict__ src16) {
  const int d = threadIdx.x;                   // 0..511
  const int p = blockIdx.x;                    // 0..31
  if (p == 0) {
    const float v = cls[d] + pos[d];
    srcf[d] = v; src16[d] = (_Float16)v;
  } else {
    srcf [(size_t)(256 + p) * DMODEL + d] = 0.0f;
    src16[(size_t)(256 + p) * DMODEL + d] = (_Float16)0.0f;
  }
}

// qkv16 = src16 @ Wqkv16^T + b ; padded rows -> 0
__global__ void k_qkv(const _Float16* __restrict__ src16, const _Float16* __restrict__ W,
                      const float* __restrict__ bias, _Float16* __restrict__ qkv) {
  const int wave = (blockIdx.x * blockDim.x + threadIdx.x) >> 5;
  const int lane = threadIdx.x & 31;
  const int MT = SPAD / 16, NT = (3 * DMODEL) / 16;
  if (wave >= MT * NT) return;
  const int tm = wave % MT, tn = wave / MT;
  const int row0 = tm * 16, col0 = tn * 16;
  v8f c = {};
  for (int k0 = 0; k0 < DMODEL; k0 += 32)
    c = wmma_step(frag_a16(src16, DMODEL, row0, k0, lane),
                  frag_bt16(W, DMODEL, k0, col0, lane), c);
  const int g = lane >> 4, n = lane & 15;
#pragma unroll
  for (int r = 0; r < 8; ++r) {
    const int row = row0 + r + 8 * g;
    const int col = col0 + n;
    const float v = (row < SEQ) ? (c[r] + bias[col]) : 0.0f;
    qkv[(size_t)row * (3 * DMODEL) + col] = (_Float16)v;
  }
}

// scores[h] = Q_h @ K_h^T * DH^-0.5 (f32) ; key cols >= SEQ masked to -1e30
__global__ void k_scores(const _Float16* __restrict__ qkv, float* __restrict__ scf) {
  const int wave = (blockIdx.x * blockDim.x + threadIdx.x) >> 5;
  const int lane = threadIdx.x & 31;
  const int T = SPAD / 16;
  if (wave >= NHEAD * T * T) return;
  const int h = wave / (T * T);
  const int t = wave % (T * T);
  const int tm = t % T, tn = t / T;
  const int row0 = tm * 16, col0 = tn * 16;
  const _Float16* Q = qkv + h * DHEAD;
  const _Float16* K = qkv + DMODEL + h * DHEAD;
  v8f c = {};
#pragma unroll
  for (int k0 = 0; k0 < DHEAD; k0 += 32)
    c = wmma_step(frag_a16(Q, 3 * DMODEL, row0, k0, lane),
                  frag_bt16(K, 3 * DMODEL, k0, col0, lane), c);
  float* out = scf + (size_t)h * SPAD * SPAD;
  const int g = lane >> 4, n = lane & 15;
#pragma unroll
  for (int r = 0; r < 8; ++r) {
    const int row = row0 + r + 8 * g;
    const int col = col0 + n;
    float v = c[r] * 0.125f;                   // DH^-0.5
    if (col >= SEQ) v = -1e30f;
    out[(size_t)row * SPAD + col] = v;
  }
}

// softmax rows of scf (f32) -> pr16 (f16 probabilities)
__global__ void k_softmax(const float* __restrict__ scf, _Float16* __restrict__ pr16) {
  __shared__ float red[256];
  const int t = threadIdx.x;
  const float* p = scf + (size_t)blockIdx.x * SPAD;
  _Float16* q = pr16 + (size_t)blockIdx.x * SPAD;
  float m = -1e30f;
  for (int i = t; i < SPAD; i += 256) m = fmaxf(m, p[i]);
  red[t] = m; __syncthreads();
  for (int s = 128; s > 0; s >>= 1) { if (t < s) red[t] = fmaxf(red[t], red[t + s]); __syncthreads(); }
  m = red[0]; __syncthreads();
  float acc = 0.0f;
  for (int i = t; i < SPAD; i += 256) acc += expf(p[i] - m);
  red[t] = acc; __syncthreads();
  for (int s = 128; s > 0; s >>= 1) { if (t < s) red[t] += red[t + s]; __syncthreads(); }
  const float inv = 1.0f / red[0];
  for (int i = t; i < SPAD; i += 256) q[i] = (_Float16)(expf(p[i] - m) * inv);
}

// ctx = attn[h](288x288) @ V_h ; B from transposed V (VT16: 512 x 288)
__global__ void k_attnv(const _Float16* __restrict__ pr16, const _Float16* __restrict__ vt16,
                        _Float16* __restrict__ ob16) {
  const int wave = (blockIdx.x * blockDim.x + threadIdx.x) >> 5;
  const int lane = threadIdx.x & 31;
  const int MT = SPAD / 16, NT = DHEAD / 16;
  if (wave >= NHEAD * MT * NT) return;
  const int h = wave / (MT * NT);
  const int t = wave % (MT * NT);
  const int tm = t % MT, tn = t / MT;
  const int row0 = tm * 16, col0 = tn * 16;
  const _Float16* A = pr16 + (size_t)h * SPAD * SPAD;
  const _Float16* VT = vt16 + (size_t)h * DHEAD * SPAD;   // row = dh, ld = SPAD
  v8f c = {};
  for (int k0 = 0; k0 < SPAD; k0 += 32)
    c = wmma_step(frag_a16(A, SPAD, row0, k0, lane),
                  frag_bt16(VT, SPAD, k0, col0, lane), c);
  const int g = lane >> 4, n = lane & 15;
#pragma unroll
  for (int r = 0; r < 8; ++r)
    ob16[(size_t)(row0 + r + 8 * g) * DMODEL + h * DHEAD + col0 + n] = (_Float16)c[r];
}

// tmp = ctx @ Wo^T + bo + srcf (residual, f32) ; padded rows -> 0
__global__ void k_oproj(const _Float16* __restrict__ ob16, const _Float16* __restrict__ Wo,
                        const float* __restrict__ bo, const float* __restrict__ srcf,
                        float* __restrict__ out) {
  const int wave = (blockIdx.x * blockDim.x + threadIdx.x) >> 5;
  const int lane = threadIdx.x & 31;
  const int MT = SPAD / 16, NT = DMODEL / 16;
  if (wave >= MT * NT) return;
  const int tm = wave % MT, tn = wave / MT;
  const int row0 = tm * 16, col0 = tn * 16;
  v8f c = {};
  for (int k0 = 0; k0 < DMODEL; k0 += 32)
    c = wmma_step(frag_a16(ob16, DMODEL, row0, k0, lane),
                  frag_bt16(Wo, DMODEL, k0, col0, lane), c);
  const int g = lane >> 4, n = lane & 15;
#pragma unroll
  for (int r = 0; r < 8; ++r) {
    const int row = row0 + r + 8 * g;
    const int col = col0 + n;
    const float v = (row < SEQ)
        ? (c[r] + bo[col] + srcf[(size_t)row * DMODEL + col]) : 0.0f;
    out[(size_t)row * DMODEL + col] = v;
  }
}

// LayerNorm rows of x (f32) -> srcf (f32) and src16 (f16)
__global__ void k_ln(const float* __restrict__ x, const float* __restrict__ gm,
                     const float* __restrict__ bt, float* __restrict__ srcf,
                     _Float16* __restrict__ src16) {
  __shared__ float red[256];
  const int t = threadIdx.x;
  const float* p = x + (size_t)blockIdx.x * DMODEL;
  const float v0 = p[t], v1 = p[t + 256];
  red[t] = v0 + v1; __syncthreads();
  for (int s = 128; s > 0; s >>= 1) { if (t < s) red[t] += red[t + s]; __syncthreads(); }
  const float mu = red[0] * (1.0f / DMODEL); __syncthreads();
  const float d0 = v0 - mu, d1 = v1 - mu;
  red[t] = d0 * d0 + d1 * d1; __syncthreads();
  for (int s = 128; s > 0; s >>= 1) { if (t < s) red[t] += red[t + s]; __syncthreads(); }
  const float inv = rsqrtf(red[0] * (1.0f / DMODEL) + LN_EPS);
  const float o0 = d0 * inv * gm[t]       + bt[t];
  const float o1 = d1 * inv * gm[t + 256] + bt[t + 256];
  float* of = srcf + (size_t)blockIdx.x * DMODEL;
  _Float16* oh = src16 + (size_t)blockIdx.x * DMODEL;
  of[t] = o0;        of[t + 256] = o1;
  oh[t] = (_Float16)o0; oh[t + 256] = (_Float16)o1;
}

// ff = relu(src16 @ W1T^T + b1)  (W1T: 2048x512 f16 = ff_w1 transposed)
__global__ void k_ff1(const _Float16* __restrict__ src16, const _Float16* __restrict__ W1T,
                      const float* __restrict__ b1, _Float16* __restrict__ ff16) {
  const int wave = (blockIdx.x * blockDim.x + threadIdx.x) >> 5;
  const int lane = threadIdx.x & 31;
  const int MT = SPAD / 16, NT = DFF / 16;
  if (wave >= MT * NT) return;
  const int tm = wave % MT, tn = wave / MT;
  const int row0 = tm * 16, col0 = tn * 16;
  v8f c = {};
  for (int k0 = 0; k0 < DMODEL; k0 += 32)
    c = wmma_step(frag_a16(src16, DMODEL, row0, k0, lane),
                  frag_bt16(W1T, DMODEL, k0, col0, lane), c);
  const int g = lane >> 4, n = lane & 15;
#pragma unroll
  for (int r = 0; r < 8; ++r) {
    const int row = row0 + r + 8 * g;
    const int col = col0 + n;
    ff16[(size_t)row * DFF + col] = (_Float16)fmaxf(c[r] + b1[col], 0.0f);
  }
}

// tmp = ff16 @ W2T^T + b2 + srcf (residual); padded rows -> 0
__global__ void k_ff2(const _Float16* __restrict__ ff16, const _Float16* __restrict__ W2T,
                      const float* __restrict__ b2, const float* __restrict__ srcf,
                      float* __restrict__ out) {
  const int wave = (blockIdx.x * blockDim.x + threadIdx.x) >> 5;
  const int lane = threadIdx.x & 31;
  const int MT = SPAD / 16, NT = DMODEL / 16;
  if (wave >= MT * NT) return;
  const int tm = wave % MT, tn = wave / MT;
  const int row0 = tm * 16, col0 = tn * 16;
  v8f c = {};
  for (int k0 = 0; k0 < DFF; k0 += 32)
    c = wmma_step(frag_a16(ff16, DFF, row0, k0, lane),
                  frag_bt16(W2T, DFF, k0, col0, lane), c);
  const int g = lane >> 4, n = lane & 15;
#pragma unroll
  for (int r = 0; r < 8; ++r) {
    const int row = row0 + r + 8 * g;
    const int col = col0 + n;
    const float v = (row < SEQ)
        ? (c[r] + b2[col] + srcf[(size_t)row * DMODEL + col]) : 0.0f;
    out[(size_t)row * DMODEL + col] = v;
  }
}

// logits = srcf[0](512) @ out_w(512x65536) + out_b  (HBM bandwidth bound, f32)
__global__ void k_outproj(const float* __restrict__ src0, const float* __restrict__ Wout,
                          const float* __restrict__ bout, float* __restrict__ logits) {
  __shared__ float s0[DMODEL];
  const int t = threadIdx.x;
  s0[t] = src0[t]; s0[t + 256] = src0[t + 256];
  __syncthreads();
  const int n = blockIdx.x * 256 + t;
  float acc = bout[n];
  for (int k = 0; k < DMODEL; ++k) {
    if (k + 8 < DMODEL)
      __builtin_prefetch(&Wout[(size_t)(k + 8) * (N_TOK * N_TOK) + n], 0, 0);
    acc += s0[k] * Wout[(size_t)k * (N_TOK * N_TOK) + n];
  }
  logits[n] = acc;
}

// -------- Sinkhorn: la -= lse(rows per column), then la -= lse(cols per row)
__global__ void k_sink_col(float* __restrict__ la) {
  __shared__ float red[256];
  const int c = blockIdx.x, t = threadIdx.x;
  const float v = la[(size_t)t * N_TOK + c];
  red[t] = v; __syncthreads();
  for (int s = 128; s > 0; s >>= 1) { if (t < s) red[t] = fmaxf(red[t], red[t + s]); __syncthreads(); }
  const float m = red[0]; __syncthreads();
  red[t] = expf(v - m); __syncthreads();
  for (int s = 128; s > 0; s >>= 1) { if (t < s) red[t] += red[t + s]; __syncthreads(); }
  la[(size_t)t * N_TOK + c] = v - (m + logf(red[0]));
}

__global__ void k_sink_row(float* __restrict__ la) {
  __shared__ float red[256];
  const int r = blockIdx.x, t = threadIdx.x;
  const float v = la[(size_t)r * N_TOK + t];
  red[t] = v; __syncthreads();
  for (int s = 128; s > 0; s >>= 1) { if (t < s) red[t] = fmaxf(red[t], red[t + s]); __syncthreads(); }
  const float m = red[0]; __syncthreads();
  red[t] = expf(v - m); __syncthreads();
  for (int s = 128; s > 0; s >>= 1) { if (t < s) red[t] += red[t + s]; __syncthreads(); }
  la[(size_t)r * N_TOK + t] = v - (m + logf(red[0]));
}

// P_hat = exp(la); P_hard = one_hot(argmax per row)
__global__ void k_final(const float* __restrict__ la, float* __restrict__ out) {
  __shared__ float rv[256];
  __shared__ int   ri[256];
  const int r = blockIdx.x, t = threadIdx.x;
  const float v = la[(size_t)r * N_TOK + t];
  out[(size_t)r * N_TOK + t] = expf(v);
  rv[t] = v; ri[t] = t; __syncthreads();
  for (int s = 128; s > 0; s >>= 1) {
    if (t < s && rv[t + s] > rv[t]) { rv[t] = rv[t + s]; ri[t] = ri[t + s]; }
    __syncthreads();
  }
  const int amax = ri[0];
  out[(size_t)(N_TOK * N_TOK) + (size_t)r * N_TOK + t] = (t == amax) ? 1.0f : 0.0f;
}

// ---------------------------------------------------------------------------

extern "C" void kernel_launch(void* const* d_in, const int* in_sizes, int n_in,
                              void* d_out, int out_size, void* d_ws, size_t ws_size,
                              hipStream_t stream) {
  (void)in_sizes; (void)n_in; (void)out_size; (void)ws_size;
  const float* in_x      = (const float*)d_in[0];
  const float* emb_w     = (const float*)d_in[1];
  const float* emb_b     = (const float*)d_in[2];
  const float* pos_emb   = (const float*)d_in[3];
  const float* cls_emb   = (const float*)d_in[4];
  const float* in_proj_w = (const float*)d_in[5];
  const float* in_proj_b = (const float*)d_in[6];
  const float* attn_o_w  = (const float*)d_in[7];
  const float* attn_o_b  = (const float*)d_in[8];
  const float* ff_w1     = (const float*)d_in[9];
  const float* ff_b1     = (const float*)d_in[10];
  const float* ff_w2     = (const float*)d_in[11];
  const float* ff_b2     = (const float*)d_in[12];
  const float* ln1_g     = (const float*)d_in[13];
  const float* ln1_b     = (const float*)d_in[14];
  const float* ln2_g     = (const float*)d_in[15];
  const float* ln2_b     = (const float*)d_in[16];
  const float* out_w     = (const float*)d_in[17];
  const float* out_b     = (const float*)d_in[18];

  // ---- workspace carve-up -------------------------------------------------
  float* wsf = (float*)d_ws;
  float* SRCF = wsf;                                   // 288*512   f32
  float* TMP  = SRCF + (size_t)SPAD * DMODEL;          // 288*512   f32
  float* SCF  = TMP  + (size_t)SPAD * DMODEL;          // 8*288*288 f32
  float* LG   = SCF  + (size_t)NHEAD * SPAD * SPAD;    // 256*256   f32
  _Float16* wsh = (_Float16*)(LG + (size_t)N_TOK * N_TOK);
  _Float16* X16   = wsh;                               // 256*64
  _Float16* WEMBT = X16   + (size_t)N_TOK * 64;        // 512*64
  _Float16* SRC16 = WEMBT + (size_t)DMODEL * 64;       // 288*512
  _Float16* QKV16 = SRC16 + (size_t)SPAD * DMODEL;     // 288*1536
  _Float16* PR16  = QKV16 + (size_t)SPAD * 3 * DMODEL; // 8*288*288
  _Float16* VT16  = PR16  + (size_t)NHEAD * SPAD * SPAD; // 512*288
  _Float16* OB16  = VT16  + (size_t)DMODEL * SPAD;     // 288*512
  _Float16* FF16  = OB16  + (size_t)SPAD * DMODEL;     // 288*2048
  _Float16* WQ16  = FF16  + (size_t)SPAD * DFF;        // 4*1536*512
  _Float16* WO16  = WQ16  + (size_t)NLAYER * 3 * DMODEL * DMODEL; // 4*512*512
  _Float16* W1T   = WO16  + (size_t)NLAYER * DMODEL * DMODEL;     // 4*2048*512
  _Float16* W2T   = W1T   + (size_t)NLAYER * DFF * DMODEL;        // 4*512*2048

  const dim3 blk(256);   // 8 wave32 waves per block

  // ---- one-time (per call) precision conversion of operands --------------
  k_cvt<<<dim3((N_TOK * 64 / 4 + 255) / 256), blk, 0, stream>>>(in_x, X16, N_TOK * 64 / 4);
  k_tr32<<<dim3(DMODEL / 32, 64 / 32), blk, 0, stream>>>(emb_w, WEMBT, 64, DMODEL);
  for (int l = 0; l < NLAYER; ++l) {
    k_cvt<<<dim3(3 * DMODEL * DMODEL / 4 / 256), blk, 0, stream>>>(
        in_proj_w + (size_t)l * 3 * DMODEL * DMODEL,
        WQ16 + (size_t)l * 3 * DMODEL * DMODEL, 3 * DMODEL * DMODEL / 4);
    k_cvt<<<dim3(DMODEL * DMODEL / 4 / 256), blk, 0, stream>>>(
        attn_o_w + (size_t)l * DMODEL * DMODEL,
        WO16 + (size_t)l * DMODEL * DMODEL, DMODEL * DMODEL / 4);
    k_tr32<<<dim3(DFF / 32, DMODEL / 32), blk, 0, stream>>>(
        ff_w1 + (size_t)l * DMODEL * DFF, W1T + (size_t)l * DFF * DMODEL, DMODEL, DFF);
    k_tr32<<<dim3(DMODEL / 32, DFF / 32), blk, 0, stream>>>(
        ff_w2 + (size_t)l * DFF * DMODEL, W2T + (size_t)l * DMODEL * DFF, DFF, DMODEL);
  }

  // ---- embed --------------------------------------------------------------
  k_embed<<<dim3((16 * 32) / 8), blk, 0, stream>>>(X16, WEMBT, emb_b, pos_emb, SRCF, SRC16);
  k_edges<<<dim3(SPAD - 256), dim3(512), 0, stream>>>(cls_emb, pos_emb, SRCF, SRC16);

  // ---- transformer layers -------------------------------------------------
  for (int l = 0; l < NLAYER; ++l) {
    const _Float16* Wqkv = WQ16 + (size_t)l * 3 * DMODEL * DMODEL;
    const _Float16* Wo   = WO16 + (size_t)l * DMODEL * DMODEL;
    const _Float16* w1t  = W1T  + (size_t)l * DFF * DMODEL;
    const _Float16* w2t  = W2T  + (size_t)l * DMODEL * DFF;
    const float* bqkv = in_proj_b + (size_t)l * 3 * DMODEL;
    const float* bo   = attn_o_b  + (size_t)l * DMODEL;
    const float* b1   = ff_b1     + (size_t)l * DFF;
    const float* b2   = ff_b2     + (size_t)l * DMODEL;

    k_qkv    <<<dim3((18 * 96) / 8),     blk, 0, stream>>>(SRC16, Wqkv, bqkv, QKV16);
    // transpose V block (rows=288 tokens, cols=512 feats) -> VT16 (512 x 288)
    k_tr16   <<<dim3(DMODEL / 32, SPAD / 32), blk, 0, stream>>>(
                  QKV16 + 2 * DMODEL, 3 * DMODEL, VT16, SPAD, DMODEL);
    k_scores <<<dim3((8 * 18 * 18) / 8), blk, 0, stream>>>(QKV16, SCF);
    k_softmax<<<dim3(NHEAD * SPAD),      blk, 0, stream>>>(SCF, PR16);
    k_attnv  <<<dim3((8 * 18 * 4) / 8),  blk, 0, stream>>>(PR16, VT16, OB16);
    k_oproj  <<<dim3((18 * 32) / 8),     blk, 0, stream>>>(OB16, Wo, bo, SRCF, TMP);
    k_ln     <<<dim3(SPAD),              blk, 0, stream>>>(TMP, ln1_g + (size_t)l * DMODEL,
                                                           ln1_b + (size_t)l * DMODEL, SRCF, SRC16);
    k_ff1    <<<dim3((18 * 128) / 8),    blk, 0, stream>>>(SRC16, w1t, b1, FF16);
    k_ff2    <<<dim3((18 * 32) / 8),     blk, 0, stream>>>(FF16, w2t, b2, SRCF, TMP);
    k_ln     <<<dim3(SPAD),              blk, 0, stream>>>(TMP, ln2_g + (size_t)l * DMODEL,
                                                           ln2_b + (size_t)l * DMODEL, SRCF, SRC16);
  }

  // ---- head + sinkhorn ----------------------------------------------------
  k_outproj<<<dim3((N_TOK * N_TOK) / 256), blk, 0, stream>>>(SRCF, out_w, out_b, LG);

  for (int it = 0; it < NITER; ++it) {
    k_sink_col<<<dim3(N_TOK), blk, 0, stream>>>(LG);
    k_sink_row<<<dim3(N_TOK), blk, 0, stream>>>(LG);
  }

  k_final<<<dim3(N_TOK), blk, 0, stream>>>(LG, (float*)d_out);
}
